// WinDecoderTransformer_71141838291582
// MI455X (gfx1250) — matrix operand order repaired and checked by reading, not compiled, not run on hardware
//
#include <hip/hip_runtime.h>
#include <hip/hip_bf16.h>
#include <math.h>

// ---------------------------------------------------------------------------
// Types / constants
// ---------------------------------------------------------------------------
typedef _Float16 h16;
typedef _Float16 v16h __attribute__((ext_vector_type(16)));
typedef _Float16 v8h  __attribute__((ext_vector_type(8)));
typedef float    v8f  __attribute__((ext_vector_type(8)));

#define HDIM   192
#define WDIM   192
#define HWTOT  (HDIM * WDIM)     // 36864 pixels == tokens
#define TTOT   HWTOT
#define CDIM   256
#define NHEADC 8
#define DHEAD  32
#define FFDIM  512
#define NLAYER 2

__device__ __forceinline__ v8f vzero8() {
  v8f z;
#pragma unroll
  for (int i = 0; i < 8; ++i) z[i] = 0.0f;
  return z;
}

// A-fragment (16x32, f16, row-major source with row stride ld).
// Lane l (0-15): row = l, K halves {0..7} and {16..23}.
// Lane l (16-31): row = l-16, K halves {8..15} and {24..31}.
__device__ __forceinline__ v16h load_frag_a(const h16* base, int ld) {
  const int l = threadIdx.x & 31;
  const h16* p = base + (size_t)(l & 15) * ld + ((l >> 4) << 3);
  v8h lo = *reinterpret_cast<const v8h*>(p);
  v8h hi = *reinterpret_cast<const v8h*>(p + 16);
  v16h r;
#pragma unroll
  for (int i = 0; i < 8; ++i) { r[i] = lo[i]; r[i + 8] = hi[i]; }
  return r;
}

// B-fragment (32x16) from W stored row-major [N,K] (so B[k][n] = W[n][k]).
// Lane l (0-15): col n = l, K halves {0..7} and {16..23}; lanes 16-31: {8..15} and {24..31}.
__device__ __forceinline__ v16h load_frag_b(const h16* base, int ld) {
  const int l = threadIdx.x & 31;
  const h16* p = base + (size_t)(l & 15) * ld + ((l >> 4) << 3);
  v8h lo = *reinterpret_cast<const v8h*>(p);
  v8h hi = *reinterpret_cast<const v8h*>(p + 16);
  v16h r;
#pragma unroll
  for (int i = 0; i < 8; ++i) { r[i] = lo[i]; r[i + 8] = hi[i]; }
  return r;
}

__device__ __forceinline__ v8f wmma16(v16h a, v16h b, v8f c) {
  return __builtin_amdgcn_wmma_f32_16x16x32_f16(false, a, false, b,
                                                (short)0, c, false, false);
}

// ---------------------------------------------------------------------------
// Generic WMMA GEMM:  out[M,N] = A[M,K](f16) @ W[N,K]^T(f16) + bias, opt ReLU.
// grid = (N/128, M/64), block = 256 (8 waves, each owns a 32x32 tile).
// trans_shift != 0: f16 output stored per-window transposed with
// lw = 1<<trans_shift:  outH[win*lw*CDIM + col*lw + (row & (lw-1))]
// (win = row >> trans_shift) so the attention P@V B-operand can use the fast
// contiguous-K fragment load.  Shift/mask only -- lw is a power of two.
// ---------------------------------------------------------------------------
__global__ void gemm_wmma_kernel(const h16* __restrict__ A, int lda,
                                 const h16* __restrict__ W, int ldw,
                                 const float* __restrict__ bias,
                                 float* __restrict__ outF,
                                 h16* __restrict__ outH, int ldo,
                                 int K, int relu, int trans_shift) {
  const int wave = threadIdx.x >> 5;
  const int lane = threadIdx.x & 31;
  const int m0 = blockIdx.y * 64 + (wave >> 2) * 32;
  const int n0 = blockIdx.x * 128 + (wave & 3) * 32;

  v8f acc[2][2];
#pragma unroll
  for (int i = 0; i < 2; ++i)
#pragma unroll
    for (int j = 0; j < 2; ++j) acc[i][j] = vzero8();

  for (int k0 = 0; k0 < K; k0 += 32) {
    // Speculative prefetch of the next A tiles (global_prefetch_b8):
    // unconditional -- OOB speculative prefetches are silently dropped,
    // keeping the inner loop branch-free.
    __builtin_prefetch(A + (size_t)m0 * lda + k0 + 32, 0, 1);
    __builtin_prefetch(A + (size_t)(m0 + 16) * lda + k0 + 32, 0, 1);
    v16h a0 = load_frag_a(A + (size_t)m0 * lda + k0, lda);
    v16h a1 = load_frag_a(A + (size_t)(m0 + 16) * lda + k0, lda);
    v16h b0 = load_frag_b(W + (size_t)n0 * ldw + k0, ldw);
    v16h b1 = load_frag_b(W + (size_t)(n0 + 16) * ldw + k0, ldw);
    acc[0][0] = wmma16(a0, b0, acc[0][0]);
    acc[0][1] = wmma16(a0, b1, acc[0][1]);
    acc[1][0] = wmma16(a1, b0, acc[1][0]);
    acc[1][1] = wmma16(a1, b1, acc[1][1]);
  }

  const int nl = lane & 15;
  const int mh = (lane >> 4) << 3;  // 0 or 8
#pragma unroll
  for (int ci = 0; ci < 2; ++ci)
#pragma unroll
    for (int cj = 0; cj < 2; ++cj) {
      const int col = n0 + cj * 16 + nl;
      const float bv = bias ? bias[col] : 0.0f;
#pragma unroll
      for (int r = 0; r < 8; ++r) {
        const int row = m0 + ci * 16 + mh + r;
        float v = acc[ci][cj][r] + bv;
        if (relu) v = v > 0.0f ? v : 0.0f;
        if (outF) outF[(size_t)row * ldo + col] = v;
        if (outH) {
          if (trans_shift) {
            const int win = row >> trans_shift;
            const int l = row & ((1 << trans_shift) - 1);
            outH[(((size_t)win * CDIM + col) << trans_shift) + l] = (h16)v;
          } else {
            outH[(size_t)row * ldo + col] = (h16)v;
          }
        }
      }
    }
}

// ---------------------------------------------------------------------------
// Fused windowed attention (per window, per head).
// Q/K are [T, 256] f16 token-major; V is per-window transposed [win][ch][tok].
// grid = (Nwin, 8), block = (LW/16) waves. S == LW.
// K tile for this (window, head) is staged to LDS with
// global_load_async_to_lds_b128 (ASYNCcnt path), then read as B-fragments.
// ---------------------------------------------------------------------------
template <int LW>
__global__ void attn_kernel(const h16* __restrict__ Qb,
                            const h16* __restrict__ Kb,
                            const h16* __restrict__ Vt,
                            const unsigned char* __restrict__ kpm,
                            h16* __restrict__ Ob, float scale) {
  constexpr int S  = LW;
  constexpr int NT = LW / 16;
  constexpr int NW = LW / 16;  // waves per block
  extern __shared__ __align__(16) char smem[];

  const int win  = blockIdx.x;
  const int head = blockIdx.y;
  const int wave = threadIdx.x >> 5;
  const int lane = threadIdx.x & 31;
  const int row0 = win * LW;
  const int co   = head * DHEAD;

  h16* probs = reinterpret_cast<h16*>(smem);      // NW * 16 * S halves
  h16* kt    = probs + (size_t)NW * 16 * S;       // LW * 32 halves
  h16* sp    = probs + (size_t)wave * 16 * S;

  const h16* qp = Qb + (size_t)(row0 + wave * 16) * CDIM + co;
  const h16* kbase = Kb + (size_t)row0 * CDIM + co;
  // transposed V for this window/head: rows = channels, cols = tokens
  const h16* vp = Vt + (size_t)win * LW * CDIM + (size_t)co * LW;

  // ---- stage K tile [LW x 32] into LDS via async copy (16B per lane) ----
  {
    const unsigned ldsbase = (unsigned)(size_t)kt;
    const int id = wave * 32 + lane;
    constexpr int NCHUNK = LW * 4;       // 16-byte chunks
    constexpr int STRIDE = NW * 32;      // chunks per round
#pragma unroll
    for (int c0 = 0; c0 < NCHUNK; c0 += STRIDE) {
      const int cid = c0 + id;
      const h16* g = kbase + (size_t)(cid >> 2) * CDIM + ((cid & 3) << 3);
      const unsigned ldst = ldsbase + (unsigned)cid * 16u;
      asm volatile("global_load_async_to_lds_b128 %0, %1, off"
                   :: "v"(ldst), "v"(g)
                   : "memory");
    }
    asm volatile("s_wait_asynccnt 0x0" ::: "memory");
  }
  __syncthreads();

  // ---- scores: one WMMA per 16x16 tile (K-dim == head dim == 32) ----
  v16h aq = load_frag_a(qp, CDIM);
  v8f sc[NT];
#pragma unroll
  for (int j = 0; j < NT; ++j) {
    v16h bk = load_frag_b(kt + (size_t)(j * 16) * 32, 32);  // from LDS
    sc[j] = wmma16(aq, bk, vzero8());
  }

  // scale + key-padding mask (mask depends only on key index)
#pragma unroll
  for (int j = 0; j < NT; ++j) {
    float madd = 0.0f;
    if (kpm && kpm[row0 + j * 16 + (lane & 15)]) madd = -1e9f;
#pragma unroll
    for (int r = 0; r < 8; ++r) sc[j][r] = sc[j][r] * scale + madd;
  }

  // softmax: C-fragment row = (lane>>4)*8 + r -> per-r reductions across the
  // 16-lane half give exact row max/sum; row sums stay in registers.
  float rs[8];
#pragma unroll
  for (int r = 0; r < 8; ++r) {
    float mx = -3.4e38f;
#pragma unroll
    for (int j = 0; j < NT; ++j) mx = fmaxf(mx, sc[j][r]);
    for (int off = 8; off; off >>= 1) mx = fmaxf(mx, __shfl_xor(mx, off, 16));
    float sum = 0.0f;
#pragma unroll
    for (int j = 0; j < NT; ++j) {
      float e = __expf(sc[j][r] - mx);
      sc[j][r] = e;
      sum += e;
    }
    for (int off = 8; off; off >>= 1) sum += __shfl_xor(sum, off, 16);
    rs[r] = sum;
  }

  // stage unnormalized probs to LDS as f16 A-matrix input
  const int ml = (lane >> 4) << 3;
#pragma unroll
  for (int j = 0; j < NT; ++j)
#pragma unroll
    for (int r = 0; r < 8; ++r)
      sp[(size_t)(ml + r) * S + j * 16 + (lane & 15)] = (h16)sc[j][r];
  __syncthreads();

  // O = P @ V : K-dim = S in steps of 32, head dim 32 = two 16-col tiles.
  // B operand comes from transposed V -> fast contiguous-K fragment loads.
  v8f oa[2];
  oa[0] = vzero8();
  oa[1] = vzero8();
#pragma unroll
  for (int k0 = 0; k0 < S; k0 += 32) {
    v16h ap = load_frag_a(sp + k0, S);
#pragma unroll
    for (int t = 0; t < 2; ++t) {
      v16h bv = load_frag_b(vp + (size_t)(t * 16) * LW + k0, LW);
      oa[t] = wmma16(ap, bv, oa[t]);
    }
  }

#pragma unroll
  for (int t = 0; t < 2; ++t)
#pragma unroll
    for (int r = 0; r < 8; ++r) {
      const int m = ml + r;
      const float v = oa[t][r] / rs[r];
      Ob[(size_t)(row0 + wave * 16 + m) * CDIM + co + t * 16 + (lane & 15)] =
          (h16)v;
    }
}

// ---------------------------------------------------------------------------
// Elementwise helpers
// ---------------------------------------------------------------------------
__global__ void f32_to_f16_kernel(const float* __restrict__ a,
                                  h16* __restrict__ o, int n) {
  int i = blockIdx.x * blockDim.x + threadIdx.x;
  if (i < n) o[i] = (h16)a[i];
}

__global__ void add_f16_kernel(const float* __restrict__ a,
                               const float* __restrict__ b,
                               h16* __restrict__ o, int n) {
  int i = blockIdx.x * blockDim.x + threadIdx.x;
  if (i < n) o[i] = (h16)(a[i] + b[i]);
}

// Wave-per-row residual + LayerNorm over C=256. block = 8 waves.
__global__ void add_ln_kernel(const float* __restrict__ res,
                              const float* __restrict__ y,
                              const float* __restrict__ g,
                              const float* __restrict__ b,
                              float* __restrict__ outF,
                              h16* __restrict__ outH) {
  const int wave = threadIdx.x >> 5;
  const int lane = threadIdx.x & 31;
  const size_t row = (size_t)blockIdx.x * 8 + wave;
  const float* yr = y + row * CDIM;
  const float* rr = res ? res + row * CDIM : nullptr;

  float v[8];
  float s = 0.0f;
#pragma unroll
  for (int i = 0; i < 8; ++i) {
    const int c = lane + i * 32;
    float x = yr[c] + (rr ? rr[c] : 0.0f);
    v[i] = x;
    s += x;
  }
  for (int off = 16; off; off >>= 1) s += __shfl_xor(s, off, 32);
  const float mean = s * (1.0f / CDIM);
  float vs = 0.0f;
#pragma unroll
  for (int i = 0; i < 8; ++i) {
    const float d = v[i] - mean;
    vs += d * d;
  }
  for (int off = 16; off; off >>= 1) vs += __shfl_xor(vs, off, 32);
  const float inv = rsqrtf(vs * (1.0f / CDIM) + 1e-5f);
#pragma unroll
  for (int i = 0; i < 8; ++i) {
    const int c = lane + i * 32;
    const float o = (v[i] - mean) * inv * g[c] + b[c];
    outF[row * CDIM + c] = o;
    outH[row * CDIM + c] = (h16)o;
  }
}

// ---------------------------------------------------------------------------
// Window (de)partition. Token row = win*Lw + l, win-major contiguous windows.
// win = (si*s + sj)*nh*nw + ih*nw + iw ; l = i*ww + j
// h = (ih*wh + i)*s + si ; w = (iw*ww + j)*s + sj
// ---------------------------------------------------------------------------
__device__ __forceinline__ int tok_to_pixel(int tok, int s, int wh, int ww) {
  const int Lw = wh * ww;
  const int win = tok / Lw, l = tok - win * Lw;
  const int nh = (HDIM / s) / wh, nw = (WDIM / s) / ww;
  const int iw = win % nw;
  int t = win / nw;
  const int ih = t % nh;
  t /= nh;
  const int sj = t % s;
  const int si = t / s;
  const int i = l / ww, j = l - i * ww;
  const int h = (ih * wh + i) * s + si;
  const int w = (iw * ww + j) * s + sj;
  return h * WDIM + w;
}

__global__ void partition_kernel(const float* __restrict__ qf_grid,
                                 const float* __restrict__ qe_grid,
                                 const float* __restrict__ src_grid,
                                 const float* __restrict__ pos_grid,
                                 const unsigned char* __restrict__ mask,
                                 float* __restrict__ t32, h16* __restrict__ t16,
                                 float* __restrict__ qpos32,
                                 h16* __restrict__ mem16,
                                 h16* __restrict__ kmem16,
                                 unsigned char* __restrict__ kpm,
                                 int s, int wh, int ww) {
  const int tok = blockIdx.x;
  const int c = threadIdx.x;
  const int g = tok_to_pixel(tok, s, wh, ww);
  const size_t gi = (size_t)c * HWTOT + g;
  const float qv = qf_grid[gi];
  const float pv = qe_grid[gi];
  const float sv = src_grid[gi];
  const float ev = pos_grid[gi];
  const size_t o = (size_t)tok * CDIM + c;
  t32[o] = qv;
  t16[o] = (h16)qv;
  qpos32[o] = pv;
  mem16[o] = (h16)sv;
  kmem16[o] = (h16)(sv + ev);
  if (c == 0) kpm[tok] = mask[g] ? 1 : 0;
}

__global__ void unpartition_kernel(const float* __restrict__ t32,
                                   float* __restrict__ grid,
                                   float* __restrict__ hs,
                                   int s, int wh, int ww) {
  const int tok = blockIdx.x;
  const int c = threadIdx.x;
  const int g = tok_to_pixel(tok, s, wh, ww);
  const float v = t32[(size_t)tok * CDIM + c];
  if (grid) grid[(size_t)c * HWTOT + g] = v;
  if (hs) hs[(size_t)g * CDIM + c] = v;  // hs: (H*W, C) token-major
}

// ---------------------------------------------------------------------------
// Host orchestration
// ---------------------------------------------------------------------------
extern "C" void kernel_launch(void* const* d_in, const int* in_sizes, int n_in,
                              void* d_out, int out_size, void* d_ws,
                              size_t ws_size, hipStream_t stream) {
  (void)in_sizes; (void)n_in; (void)out_size; (void)ws_size;

  const float* src      = (const float*)d_in[0];
  const float* pos      = (const float*)d_in[1];
  const unsigned char* mask = (const unsigned char*)d_in[2];
  const float* qf0      = (const float*)d_in[3];
  const float* qe       = (const float*)d_in[4];
  const int*   pts      = (const int*)d_in[5];
  const float* sa_in_w  = (const float*)d_in[6];
  const float* sa_in_b  = (const float*)d_in[7];
  const float* sa_out_w = (const float*)d_in[8];
  const float* sa_out_b = (const float*)d_in[9];
  const float* ca_in_w  = (const float*)d_in[10];
  const float* ca_in_b  = (const float*)d_in[11];
  const float* ca_out_w = (const float*)d_in[12];
  const float* ca_out_b = (const float*)d_in[13];
  const float* ff1_w    = (const float*)d_in[14];
  const float* ff1_b    = (const float*)d_in[15];
  const float* ff2_w    = (const float*)d_in[16];
  const float* ff2_b    = (const float*)d_in[17];
  const float* n1_g = (const float*)d_in[18];
  const float* n1_b = (const float*)d_in[19];
  const float* n2_g = (const float*)d_in[20];
  const float* n2_b = (const float*)d_in[21];
  const float* n3_g = (const float*)d_in[22];
  const float* n3_b = (const float*)d_in[23];
  const float* dn_g = (const float*)d_in[24];
  const float* dn_b = (const float*)d_in[25];

  // --- workspace carve-out (256B aligned) ---
  char* ws = (char*)d_ws;
  size_t off = 0;
  auto carve = [&](size_t bytes) -> void* {
    void* p = ws + off;
    off = (off + bytes + 255) & ~(size_t)255;
    return p;
  };
  h16* w_sa_in  = (h16*)carve((size_t)NLAYER * 768 * 256 * 2);
  h16* w_sa_out = (h16*)carve((size_t)NLAYER * 256 * 256 * 2);
  h16* w_ca_in  = (h16*)carve((size_t)NLAYER * 768 * 256 * 2);
  h16* w_ca_out = (h16*)carve((size_t)NLAYER * 256 * 256 * 2);
  h16* w_ff1    = (h16*)carve((size_t)NLAYER * 512 * 256 * 2);
  h16* w_ff2    = (h16*)carve((size_t)NLAYER * 256 * 512 * 2);
  float* gridb  = (float*)carve((size_t)CDIM * HWTOT * 4);
  float* t32    = (float*)carve((size_t)TTOT * CDIM * 4);
  h16*   t16    = (h16*)carve((size_t)TTOT * CDIM * 2);
  float* qpos32 = (float*)carve((size_t)TTOT * CDIM * 4);
  h16*   q16    = (h16*)carve((size_t)TTOT * CDIM * 2);
  h16*   mem16  = (h16*)carve((size_t)TTOT * CDIM * 2);
  h16*   kmem16 = (h16*)carve((size_t)TTOT * CDIM * 2);
  h16*   Qb     = (h16*)carve((size_t)TTOT * CDIM * 2);
  h16*   Kb     = (h16*)carve((size_t)TTOT * CDIM * 2);
  h16*   Vt     = (h16*)carve((size_t)TTOT * CDIM * 2);  // window-transposed V
  h16*   attn16 = (h16*)carve((size_t)TTOT * CDIM * 2);
  float* proj32 = (float*)carve((size_t)TTOT * CDIM * 4);
  h16*   ffh16  = (h16*)carve((size_t)TTOT * FFDIM * 2);
  unsigned char* kpm = (unsigned char*)carve(TTOT);

  auto cvt = [&](const float* s, h16* d, int n) {
    f32_to_f16_kernel<<<(n + 255) / 256, 256, 0, stream>>>(s, d, n);
  };
  cvt(sa_in_w,  w_sa_in,  NLAYER * 768 * 256);
  cvt(sa_out_w, w_sa_out, NLAYER * 256 * 256);
  cvt(ca_in_w,  w_ca_in,  NLAYER * 768 * 256);
  cvt(ca_out_w, w_ca_out, NLAYER * 256 * 256);
  cvt(ff1_w,    w_ff1,    NLAYER * 512 * 256);
  cvt(ff2_w,    w_ff2,    NLAYER * 256 * 512);

  auto gemm_grid = [](int N) { return dim3(N / 128, TTOT / 64); };
  const int   ewGrid = (TTOT * CDIM) / 256;
  const float scale = 1.0f / sqrtf((float)DHEAD);

  const int sL[2] = {2, 1}, whL[2] = {16, 8}, wwL[2] = {8, 4};
  const int shL[2] = {7, 5};  // log2(window length): 128 -> 7, 32 -> 5

  for (int st = 0; st < 2; ++st) {
    const int s = sL[st], wh = whL[st], ww = wwL[st];
    const int Lw = wh * ww;
    const int lwShift = shL[st];
    const int Nwin = TTOT / Lw;
    const float* qgrid = (st == 0) ? qf0 : gridb;

    partition_kernel<<<TTOT, 256, 0, stream>>>(qgrid, qe, src, pos, mask, t32,
                                               t16, qpos32, mem16, kmem16, kpm,
                                               s, wh, ww);

    auto run_attn = [&](const unsigned char* m) {
      dim3 ga(Nwin, NHEADC);
      if (Lw == 128) {
        const size_t shm = (size_t)(128 / 16) * 16 * 128 * 2  // probs
                         + (size_t)128 * 32 * 2;              // K tile
        attn_kernel<128><<<ga, (128 / 16) * 32, shm, stream>>>(
            Qb, Kb, Vt, m, attn16, scale);
      } else {
        const size_t shm = (size_t)(32 / 16) * 16 * 32 * 2
                         + (size_t)32 * 32 * 2;
        attn_kernel<32><<<ga, (32 / 16) * 32, shm, stream>>>(
            Qb, Kb, Vt, m, attn16, scale);
      }
    };

    for (int li = 0; li < NLAYER; ++li) {
      // ---------------- self-attention ----------------
      const h16* wq = w_sa_in + (size_t)li * 768 * 256;
      const float* bq = sa_in_b + (size_t)li * 768;
      add_f16_kernel<<<ewGrid, 256, 0, stream>>>(t32, qpos32, q16, TTOT * CDIM);
      gemm_wmma_kernel<<<gemm_grid(256), 256, 0, stream>>>(
          q16, 256, wq, 256, bq, nullptr, Qb, 256, 256, 0, 0);
      gemm_wmma_kernel<<<gemm_grid(256), 256, 0, stream>>>(
          q16, 256, wq + 256 * 256, 256, bq + 256, nullptr, Kb, 256, 256, 0, 0);
      gemm_wmma_kernel<<<gemm_grid(256), 256, 0, stream>>>(
          t16, 256, wq + 512 * 256, 256, bq + 512, nullptr, Vt, 256, 256, 0,
          lwShift);
      run_attn(nullptr);
      gemm_wmma_kernel<<<gemm_grid(256), 256, 0, stream>>>(
          attn16, 256, w_sa_out + (size_t)li * 256 * 256, 256,
          sa_out_b + (size_t)li * 256, proj32, nullptr, 256, 256, 0, 0);
      add_ln_kernel<<<TTOT / 8, 256, 0, stream>>>(
          t32, proj32, n1_g + li * 256, n1_b + li * 256, t32, t16);

      // ---------------- cross-attention ----------------
      const h16* cwq = w_ca_in + (size_t)li * 768 * 256;
      const float* cbq = ca_in_b + (size_t)li * 768;
      add_f16_kernel<<<ewGrid, 256, 0, stream>>>(t32, qpos32, q16, TTOT * CDIM);
      gemm_wmma_kernel<<<gemm_grid(256), 256, 0, stream>>>(
          q16, 256, cwq, 256, cbq, nullptr, Qb, 256, 256, 0, 0);
      gemm_wmma_kernel<<<gemm_grid(256), 256, 0, stream>>>(
          kmem16, 256, cwq + 256 * 256, 256, cbq + 256, nullptr, Kb, 256, 256,
          0, 0);
      gemm_wmma_kernel<<<gemm_grid(256), 256, 0, stream>>>(
          mem16, 256, cwq + 512 * 256, 256, cbq + 512, nullptr, Vt, 256, 256,
          0, lwShift);
      run_attn(kpm);
      gemm_wmma_kernel<<<gemm_grid(256), 256, 0, stream>>>(
          attn16, 256, w_ca_out + (size_t)li * 256 * 256, 256,
          ca_out_b + (size_t)li * 256, proj32, nullptr, 256, 256, 0, 0);
      add_ln_kernel<<<TTOT / 8, 256, 0, stream>>>(
          t32, proj32, n2_g + li * 256, n2_b + li * 256, t32, t16);

      // ---------------- FFN ----------------
      gemm_wmma_kernel<<<gemm_grid(512), 256, 0, stream>>>(
          t16, 256, w_ff1 + (size_t)li * 512 * 256, 256,
          ff1_b + (size_t)li * 512, nullptr, ffh16, 512, 256, 1 /*relu*/, 0);
      gemm_wmma_kernel<<<gemm_grid(256), 256, 0, stream>>>(
          ffh16, 512, w_ff2 + (size_t)li * 256 * 512, 512,
          ff2_b + (size_t)li * 256, proj32, nullptr, 256, 512, 0, 0);
      add_ln_kernel<<<TTOT / 8, 256, 0, stream>>>(
          t32, proj32, n3_g + li * 256, n3_b + li * 256, t32, t16);
    }

    // decoder final LayerNorm (per stage)
    add_ln_kernel<<<TTOT / 8, 256, 0, stream>>>(nullptr, t32, dn_g, dn_b, t32,
                                                t16);

    unpartition_kernel<<<TTOT, 256, 0, stream>>>(
        t32, (st == 0) ? gridb : nullptr,
        (st == 1) ? (float*)d_out : nullptr, s, wh, ww);
  }

  // append points_queries (int32 bits) after hs
  hipMemcpyAsync((char*)d_out + (size_t)TTOT * CDIM * sizeof(float), pts,
                 (size_t)TTOT * 2 * sizeof(int), hipMemcpyDeviceToDevice,
                 stream);
}
